// EncoderLayer_57647051047503
// MI455X (gfx1250) — matrix operand order repaired and checked by reading, not compiled, not run on hardware
//
#include <hip/hip_runtime.h>
#include <math.h>

// ---------------------------------------------------------------------------
// Types for CDNA5 WMMA (wave32, 16x16x32 bf16 -> f32 accumulate)
// ---------------------------------------------------------------------------
typedef __attribute__((ext_vector_type(16))) __bf16 v16bf;
typedef __attribute__((ext_vector_type(8)))  float  v8f;

__device__ __forceinline__ unsigned short f2bf(float f) {
    unsigned u = __float_as_uint(f);
    u += 0x7FFFu + ((u >> 16) & 1u);      // round-to-nearest-even
    return (unsigned short)(u >> 16);
}

__device__ __forceinline__ v16bf make_frag(uint4 a, uint4 b) {
    union { uint4 u[2]; v16bf v; } x;
    x.u[0] = a; x.u[1] = b;
    return x.v;
}

// A fragment: 16x32 bf16, A row-major (M x K), per-lane 2 x b128 loads.
// lanes 0-15: VGPR0-3 = K 0..7, VGPR4-7 = K 16..23 ; lanes 16-31: +8
__device__ __forceinline__ v16bf loadA_g(const unsigned short* A, int lda,
                                         int row0, int col0, int lane) {
    int m = lane & 15, half = lane >> 4;
    const unsigned short* p = A + (size_t)(row0 + m) * lda + col0;
    return make_frag(*(const uint4*)(p + 8 * half),
                     *(const uint4*)(p + 16 + 8 * half));
}

// B fragment: 32x16 bf16, from BT stored row-major (N x K): 2 x b128 loads.
// lanes 0-15: VGPR0-3 = K 0..7, VGPR4-7 = K 16..23 ; lanes 16-31: +8
__device__ __forceinline__ v16bf loadB_g(const unsigned short* BT, int ldb,
                                         int col0, int k0, int lane) {
    int n = lane & 15, half = lane >> 4;
    const unsigned short* p = BT + (size_t)(col0 + n) * ldb + k0;
    return make_frag(*(const uint4*)(p + 8 * half), *(const uint4*)(p + 16 + 8 * half));
}

__device__ __forceinline__ v8f wmma_bf16(v16bf a, v16bf b, v8f c) {
    return __builtin_amdgcn_wmma_f32_16x16x32_bf16(
        false, a, false, b, (short)0, c, false, false);
}

#define ZERO8F ((v8f){0.f,0.f,0.f,0.f,0.f,0.f,0.f,0.f})

// ---------------------------------------------------------------------------
// Weight convert+transpose: in f32 [K][N]  ->  out bf16 [N][K]
// ---------------------------------------------------------------------------
__global__ void tconv_kernel(const float* __restrict__ in,
                             unsigned short* __restrict__ out,
                             int K, int N, int total) {
    int gid = blockIdx.x * blockDim.x + threadIdx.x;
    if (gid >= total) return;
    int n = gid / K, k = gid - n * K;
    out[gid] = f2bf(in[(size_t)k * N + n]);
}

// ---------------------------------------------------------------------------
// LayerNorm over 1024 cols, one block (256 thr) per row; bf16 output.
// ---------------------------------------------------------------------------
__global__ __launch_bounds__(256)
void ln_kernel(const float* __restrict__ x, const float* __restrict__ g,
               const float* __restrict__ b, unsigned short* __restrict__ out) {
    __shared__ float s1[256], s2[256];
    int row = blockIdx.x, tid = threadIdx.x;
    const float* xr = x + (size_t)row * 1024;
    float xv[4], sum = 0.f, sq = 0.f;
    for (int j = 0; j < 4; ++j) {
        xv[j] = xr[tid + j * 256];
        sum += xv[j]; sq += xv[j] * xv[j];
    }
    s1[tid] = sum; s2[tid] = sq;
    __syncthreads();
    for (int st = 128; st > 0; st >>= 1) {
        if (tid < st) { s1[tid] += s1[tid + st]; s2[tid] += s2[tid + st]; }
        __syncthreads();
    }
    float mean = s1[0] * (1.f / 1024.f);
    float var  = s2[0] * (1.f / 1024.f) - mean * mean;
    float rstd = rsqrtf(var + 1e-5f);
    unsigned short* orow = out + (size_t)row * 1024;
    for (int j = 0; j < 4; ++j) {
        int c = tid + j * 256;
        orow[c] = f2bf((xv[j] - mean) * rstd * g[c] + b[c]);
    }
}

// ---------------------------------------------------------------------------
// Generic bf16 WMMA GEMM:  C(MxN) = A(MxK,row-major bf16) * BT(NxK)^T
//   + bias, + residual(f32), optional ReLU; outputs f32 and/or bf16.
// Block = 256 thr = 8 waves; wave tile 32x32 (2x2 WMMA tiles);
// block tile 128(M) x 64(N). vtMode: scatter bf16 out as V^T [b][h][d][s].
// ---------------------------------------------------------------------------
__global__ __launch_bounds__(256)
void gemm_kernel(const unsigned short* __restrict__ A,
                 const unsigned short* __restrict__ BT,
                 const float* __restrict__ bias,
                 const float* __restrict__ residual,
                 float* __restrict__ outF, unsigned short* __restrict__ outB,
                 int M, int N, int K, float outScale, int relu, int vtMode) {
    int lane = threadIdx.x & 31, wave = threadIdx.x >> 5;
    int mw = wave & 3, nw = wave >> 2;
    int tileM = blockIdx.y * 128 + mw * 32;
    int tileN = blockIdx.x * 64  + nw * 32;

    v8f acc[2][2];
    acc[0][0] = ZERO8F; acc[0][1] = ZERO8F; acc[1][0] = ZERO8F; acc[1][1] = ZERO8F;

    for (int k0 = 0; k0 < K; k0 += 32) {
        v16bf a0 = loadA_g(A, K, tileM,      k0, lane);
        v16bf a1 = loadA_g(A, K, tileM + 16, k0, lane);
        v16bf b0 = loadB_g(BT, K, tileN,      k0, lane);
        v16bf b1 = loadB_g(BT, K, tileN + 16, k0, lane);
        acc[0][0] = wmma_bf16(a0, b0, acc[0][0]);
        acc[0][1] = wmma_bf16(a0, b1, acc[0][1]);
        acc[1][0] = wmma_bf16(a1, b0, acc[1][0]);
        acc[1][1] = wmma_bf16(a1, b1, acc[1][1]);
    }

    int n = lane & 15, half = lane >> 4;
    for (int mt = 0; mt < 2; ++mt)
        for (int nt = 0; nt < 2; ++nt)
            for (int r = 0; r < 8; ++r) {
                int row = tileM + mt * 16 + r + 8 * half;
                int col = tileN + nt * 16 + n;
                float v = acc[mt][nt][r];
                if (bias)     v += bias[col];
                if (residual) v += residual[(size_t)row * N + col];
                if (relu)     v = fmaxf(v, 0.f);
                if (outF)     outF[(size_t)row * N + col] = v;
                if (outB) {
                    unsigned short bv = f2bf(v * outScale);
                    if (vtMode) {
                        int b = row >> 11, s = row & 2047;
                        int h = col >> 6,  d = col & 63;
                        outB[(size_t)((b * 16 + h) * 64 + d) * 2048 + s] = bv;
                    } else {
                        outB[(size_t)row * N + col] = bv;
                    }
                }
            }
}

// ---------------------------------------------------------------------------
// Flash attention (fused, online softmax). One wave per 16-query tile.
//   Q: [8192][1024] bf16 (pre-scaled by 1/8), K: [8192][1024] bf16,
//   Vt: [b*16+h][64][2048] bf16, O: [8192][1024] bf16.
// Per 64-col s-chunk: 8 WMMA for S, softmax (butterfly shfl within
// 16-lane row groups), P -> LDS (bf16), 8 WMMA for O accumulate.
// ---------------------------------------------------------------------------
__global__ __launch_bounds__(32)
void attn_kernel(const unsigned short* __restrict__ Q,
                 const unsigned short* __restrict__ Kb,
                 const unsigned short* __restrict__ Vt,
                 unsigned short* __restrict__ O) {
    const int PST = 72;                       // padded LDS row stride (16B-aligned)
    __shared__ unsigned short Pl[16 * PST];

    int lane = threadIdx.x & 31;
    int n = lane & 15, half = lane >> 4, m = n;
    int qt = blockIdx.x;                      // 0..127
    int bh = blockIdx.y;                      // b*16+h, 0..63
    int b = bh >> 4, h = bh & 15;
    int qrow0 = b * 2048 + qt * 16;

    v16bf aq0 = loadA_g(Q, 1024, qrow0, h * 64,      lane);
    v16bf aq1 = loadA_g(Q, 1024, qrow0, h * 64 + 32, lane);

    v8f o[4]; o[0] = ZERO8F; o[1] = ZERO8F; o[2] = ZERO8F; o[3] = ZERO8F;
    float mrow[8], lrow[8];
    for (int r = 0; r < 8; ++r) { mrow[r] = -1e30f; lrow[r] = 0.f; }

    for (int sc = 0; sc < 2048; sc += 64) {
        // ----- S = (Q/8) K^T, 4 tiles of 16 cols -----
        v8f s[4];
        for (int nt = 0; nt < 4; ++nt) {
            int srow0 = b * 2048 + sc + nt * 16;
            v8f t = wmma_bf16(aq0, loadB_g(Kb, 1024, srow0, h * 64,      lane), ZERO8F);
            s[nt]  = wmma_bf16(aq1, loadB_g(Kb, 1024, srow0, h * 64 + 32, lane), t);
        }
        // ----- online softmax (rows live in 16-lane groups) -----
        float mnew[8], fac[8];
        for (int r = 0; r < 8; ++r) {
            float mx = fmaxf(fmaxf(s[0][r], s[1][r]), fmaxf(s[2][r], s[3][r]));
            mx = fmaxf(mx, __shfl_xor(mx, 1));
            mx = fmaxf(mx, __shfl_xor(mx, 2));
            mx = fmaxf(mx, __shfl_xor(mx, 4));
            mx = fmaxf(mx, __shfl_xor(mx, 8));
            mnew[r] = fmaxf(mrow[r], mx);
            fac[r]  = __expf(mrow[r] - mnew[r]);
            mrow[r] = mnew[r];
        }
        for (int dt = 0; dt < 4; ++dt)
            for (int r = 0; r < 8; ++r) o[dt][r] *= fac[r];
        for (int r = 0; r < 8; ++r) {
            float rs = 0.f;
            for (int nt = 0; nt < 4; ++nt) {
                float p = __expf(s[nt][r] - mnew[r]);
                s[nt][r] = p; rs += p;
            }
            rs += __shfl_xor(rs, 1);
            rs += __shfl_xor(rs, 2);
            rs += __shfl_xor(rs, 4);
            rs += __shfl_xor(rs, 8);
            lrow[r] = lrow[r] * fac[r] + rs;
        }
        // ----- P (C-layout) -> LDS row-major bf16 [16][64] -----
        __syncthreads();
        for (int nt = 0; nt < 4; ++nt)
            for (int r = 0; r < 8; ++r)
                Pl[(r + 8 * half) * PST + nt * 16 + n] = f2bf(s[nt][r]);
        __syncthreads();
        // ----- O += P(16x64) x V(64x64) -----
        for (int ks = 0; ks < 2; ++ks) {
            const unsigned short* pp = &Pl[m * PST + ks * 32];
            v16bf ap = make_frag(*(const uint4*)(pp + 8 * half),
                                 *(const uint4*)(pp + 16 + 8 * half));
            for (int dt = 0; dt < 4; ++dt) {
                v16bf bv = loadB_g(Vt, 2048, bh * 64 + dt * 16, sc + ks * 32, lane);
                o[dt] = wmma_bf16(ap, bv, o[dt]);
            }
        }
    }
    // ----- normalize and write attn bf16 [token][h*64+d] -----
    for (int dt = 0; dt < 4; ++dt)
        for (int r = 0; r < 8; ++r) {
            int row = qrow0 + r + 8 * half;
            int col = h * 64 + dt * 16 + n;
            O[(size_t)row * 1024 + col] = f2bf(o[dt][r] / lrow[r]);
        }
}

// ---------------------------------------------------------------------------
// logits = x[:,0] @ wlr + blr ; softplus losses. One block, 128 threads.
// ---------------------------------------------------------------------------
__device__ __forceinline__ float softplus(float x) {
    return fmaxf(x, 0.f) + log1pf(__expf(-fabsf(x)));
}

__global__ __launch_bounds__(128)
void loss_kernel(const float* __restrict__ xf, const float* __restrict__ wlr,
                 const float* __restrict__ blr, float* __restrict__ outL) {
    __shared__ float logits[4];
    int tid = threadIdx.x, wave = tid >> 5, lane = tid & 31;
    float sum = 0.f;
    for (int i = lane; i < 1024; i += 32)
        sum += xf[(size_t)wave * 2048 * 1024 + i] * wlr[i];
    sum += __shfl_xor(sum, 1);
    sum += __shfl_xor(sum, 2);
    sum += __shfl_xor(sum, 4);
    sum += __shfl_xor(sum, 8);
    sum += __shfl_xor(sum, 16);
    if (lane == 0) logits[wave] = sum + blr[0];
    __syncthreads();
    if (tid == 0) {
        float lr = 0.5f * (softplus(-logits[0]) + softplus(-logits[1]));
        float lf = 0.5f * (softplus( logits[2]) + softplus( logits[3]));
        outL[0] = 0.5f * (lf + lr);
        outL[1] = lf;
    }
}

// ---------------------------------------------------------------------------
// Host launcher
// ---------------------------------------------------------------------------
extern "C" void kernel_launch(void* const* d_in, const int* in_sizes, int n_in,
                              void* d_out, int out_size, void* d_ws, size_t ws_size,
                              hipStream_t stream) {
    (void)in_sizes; (void)n_in; (void)out_size; (void)ws_size;
    const float* x     = (const float*)d_in[0];
    const float* wq    = (const float*)d_in[1];
    const float* bq    = (const float*)d_in[2];
    const float* wk    = (const float*)d_in[3];
    const float* bk    = (const float*)d_in[4];
    const float* wv    = (const float*)d_in[5];
    const float* bv    = (const float*)d_in[6];
    const float* wo    = (const float*)d_in[7];
    const float* bo    = (const float*)d_in[8];
    const float* ln1g  = (const float*)d_in[9];
    const float* ln1b  = (const float*)d_in[10];
    const float* ln2g  = (const float*)d_in[11];
    const float* ln2b  = (const float*)d_in[12];
    const float* w1    = (const float*)d_in[13];
    const float* b1    = (const float*)d_in[14];
    const float* w2    = (const float*)d_in[15];
    const float* b2    = (const float*)d_in[16];
    const float* wlr   = (const float*)d_in[17];
    const float* blr   = (const float*)d_in[18];

    const int M = 4 * 2048;          // 8192 tokens
    const int D = 1024, F = 4096;

    // carve workspace (256-byte aligned chunks)
    char* ws = (char*)d_ws; size_t off = 0;
    auto carve = [&](size_t bytes) {
        void* p = ws + off;
        off += (bytes + 255) & ~(size_t)255;
        return p;
    };
    unsigned short* wqT = (unsigned short*)carve((size_t)D * D * 2);
    unsigned short* wkT = (unsigned short*)carve((size_t)D * D * 2);
    unsigned short* wvT = (unsigned short*)carve((size_t)D * D * 2);
    unsigned short* woT = (unsigned short*)carve((size_t)D * D * 2);
    unsigned short* w1T = (unsigned short*)carve((size_t)D * F * 2);
    unsigned short* w2T = (unsigned short*)carve((size_t)F * D * 2);
    unsigned short* xn  = (unsigned short*)carve((size_t)M * D * 2);
    unsigned short* qb  = (unsigned short*)carve((size_t)M * D * 2);
    unsigned short* kb  = (unsigned short*)carve((size_t)M * D * 2);
    unsigned short* vt  = (unsigned short*)carve((size_t)M * D * 2);
    unsigned short* at  = (unsigned short*)carve((size_t)M * D * 2);
    float*          x1  = (float*)         carve((size_t)M * D * 4);
    unsigned short* xn2 = (unsigned short*)carve((size_t)M * D * 2);
    unsigned short* hb  = (unsigned short*)carve((size_t)M * F * 2);
    float* xout = (float*)d_out;
    float* lossp = xout + (size_t)M * D;

    // 1) weights -> bf16 transposed
    auto tc = [&](const float* in, unsigned short* out, int K, int N) {
        int total = K * N;
        tconv_kernel<<<(total + 255) / 256, 256, 0, stream>>>(in, out, K, N, total);
    };
    tc(wq, wqT, D, D); tc(wk, wkT, D, D); tc(wv, wvT, D, D); tc(wo, woT, D, D);
    tc(w1, w1T, D, F); tc(w2, w2T, F, D);

    // 2) LN1
    ln_kernel<<<M, 256, 0, stream>>>(x, ln1g, ln1b, xn);

    // 3) Q/K/V projections (Q pre-scaled by 1/sqrt(64); V stored transposed)
    dim3 g1024(D / 64, M / 128);
    gemm_kernel<<<g1024, 256, 0, stream>>>(xn, wqT, bq, nullptr, nullptr, qb,
                                           M, D, D, 0.125f, 0, 0);
    gemm_kernel<<<g1024, 256, 0, stream>>>(xn, wkT, bk, nullptr, nullptr, kb,
                                           M, D, D, 1.0f, 0, 0);
    gemm_kernel<<<g1024, 256, 0, stream>>>(xn, wvT, bv, nullptr, nullptr, vt,
                                           M, D, D, 1.0f, 0, 1);

    // 4) fused flash attention
    attn_kernel<<<dim3(2048 / 16, 4 * 16), 32, 0, stream>>>(qb, kb, vt, at);

    // 5) output projection + residual -> x1 (f32)
    gemm_kernel<<<g1024, 256, 0, stream>>>(at, woT, bo, x, x1, nullptr,
                                           M, D, D, 1.0f, 0, 0);

    // 6) LN2
    ln_kernel<<<M, 256, 0, stream>>>(x1, ln2g, ln2b, xn2);

    // 7) FFN1 (ReLU, bf16 out)
    gemm_kernel<<<dim3(F / 64, M / 128), 256, 0, stream>>>(
        xn2, w1T, b1, nullptr, nullptr, hb, M, F, D, 1.0f, 1, 0);

    // 8) FFN2 + residual -> final x straight into d_out
    gemm_kernel<<<g1024, 256, 0, stream>>>(hb, w2T, b2, x1, xout, nullptr,
                                           M, D, F, 1.0f, 0, 0);

    // 9) logits + losses
    loss_kernel<<<1, 128, 0, stream>>>(xout, wlr, blr, lossp);
}
